// MedicalVQAModel_39943195853563
// MI455X (gfx1250) — compile-verified, matrix operand order checked by reading, not run on hardware
//
#include <hip/hip_runtime.h>
#include <math.h>
#include <stdint.h>

typedef __attribute__((ext_vector_type(16))) __bf16 v16bf;
typedef __attribute__((ext_vector_type(8)))  __bf16 v8bf;
typedef __attribute__((ext_vector_type(8)))  float  v8f;
typedef unsigned short u16;

constexpr int N = 16384, D = 1536, H = 768, E = 8, TOPK = 2;
constexpr int TILE_M = 128;                 // block rows (segment rows)
constexpr int TILE_NB = 256;                // block cols (H)
constexpr int TILE_K = 32;
constexpr int MAX_TILES = (N * TOPK) / TILE_M + E;   // 264 (256 real + <=8 padding tiles)
constexpr int MAX_ROWS  = MAX_TILES * TILE_M;        // 33792
constexpr int LDS_STRIDE = TILE_K + 8;      // 40 elems -> 80B row stride: 16B aligned, bank-conflict-free

// ---------------- helpers ----------------

__device__ inline u16 f2bf(float f) {       // round-to-nearest-even f32 -> bf16
  union { float f; unsigned u; } v; v.f = f;
  unsigned r = v.u + 0x7FFFu + ((v.u >> 16) & 1u);
  return (u16)(r >> 16);
}

__device__ inline float gelu_exact(float v) {
  return 0.5f * v * (1.0f + erff(v * 0.70710678118654752440f));
}

// CDNA5 async global->LDS copy (VGLOBAL op 98, tracked by ASYNCcnt).
// VDST VGPR holds the per-lane LDS byte address; generic pointers to __shared__
// are {aperture_hi, lds_offset}, so the low 32 bits are exactly that address.
__device__ inline void async_copy_b128(void* lds_dst, const void* gsrc) {
  const uint32_t lds_addr = (uint32_t)(uintptr_t)lds_dst;
  asm volatile("global_load_async_to_lds_b128 %0, %1, off"
               :: "v"(lds_addr), "v"(gsrc) : "memory");
}
__device__ inline void wait_asynccnt0() {
  asm volatile("s_wait_asynccnt 0x0" ::: "memory");
}

// One 16x32 bf16 operand fragment (A row-major or B column-major-in-LDS).
// ISA 7.12.2 16-bit layout: lanes 0-15 hold K=0..7 then 16..23; lanes 16-31
// hold K=8..15 then 24..31 -> two aligned b128 LDS loads per lane.
__device__ inline v16bf load_frag(const u16* base) {
  const v8bf lo = *(const v8bf*)(base);
  const v8bf hi = *(const v8bf*)(base + 16);
  v16bf r;
#pragma unroll
  for (int i = 0; i < 8; ++i) { r[i] = lo[i]; r[i + 8] = hi[i]; }
  return r;
}

// ---------------- kernel 1a: x f32 -> bf16 (layout kept [N][D]) ----------------

__global__ void convert_bf16_kernel(const float* __restrict__ src,
                                    u16* __restrict__ dst, int n) {
  int i = blockIdx.x * blockDim.x + threadIdx.x;
  int stride = gridDim.x * blockDim.x;
  for (; i < n; i += stride) dst[i] = f2bf(src[i]);
}

// ---------------- kernel 1b: expert_w [E][D][H] f32 -> [E][H][D] bf16 ----------------

__global__ __launch_bounds__(256)
void transpose_w_kernel(const float* __restrict__ ew, u16* __restrict__ wbT) {
  __shared__ float tile[32][33];
  const int e  = blockIdx.z;
  const int d0 = blockIdx.y * 32;
  const int h0 = blockIdx.x * 32;
#pragma unroll
  for (int i = threadIdx.y; i < 32; i += 8)
    tile[i][threadIdx.x] = ew[((size_t)e * D + d0 + i) * H + h0 + threadIdx.x];
  __syncthreads();
#pragma unroll
  for (int i = threadIdx.y; i < 32; i += 8)
    wbT[((size_t)e * H + h0 + i) * D + d0 + threadIdx.x] = f2bf(tile[threadIdx.x][i]);
}

// ---------------- kernel 2: router (one wave32 per token) ----------------

__global__ __launch_bounds__(256)
void router_kernel(const float* __restrict__ x, const float* __restrict__ rw,
                   const float* __restrict__ rb, int* __restrict__ tok_e,
                   float* __restrict__ tok_w, int* __restrict__ cnt) {
  const int lane = threadIdx.x & 31;
  const int wave = threadIdx.x >> 5;
  const int n = blockIdx.x * 8 + wave;
  if (n >= N) return;

  float s[E];
#pragma unroll
  for (int e = 0; e < E; ++e) s[e] = 0.0f;

  const float* xr = x + (size_t)n * D;
  for (int d = lane; d < D; d += 32) {
    const float xv = xr[d];
#pragma unroll
    for (int e = 0; e < E; ++e) s[e] += xv * rw[d * E + e];
  }
#pragma unroll
  for (int e = 0; e < E; ++e)
    for (int m = 16; m > 0; m >>= 1) s[e] += __shfl_xor(s[e], m, 32);

  if (lane == 0) {
    float mx = -1e30f;
#pragma unroll
    for (int e = 0; e < E; ++e) { s[e] += rb[e]; mx = fmaxf(mx, s[e]); }
    float p[E];
#pragma unroll
    for (int e = 0; e < E; ++e) p[e] = expf(s[e] - mx);
    int i0 = 0;
#pragma unroll
    for (int e = 1; e < E; ++e) if (p[e] > p[i0]) i0 = e;
    int i1 = (i0 == 0) ? 1 : 0;
#pragma unroll
    for (int e = 0; e < E; ++e) if (e != i0 && p[e] > p[i1]) i1 = e;
    const float inv = 1.0f / (p[i0] + p[i1]);   // top-2 renormalization
    tok_e[2 * n + 0] = i0;  tok_w[2 * n + 0] = p[i0] * inv;
    tok_e[2 * n + 1] = i1;  tok_w[2 * n + 1] = p[i1] * inv;
    atomicAdd(&cnt[i0], 1);
    atomicAdd(&cnt[i1], 1);
  }
}

// ---------------- kernel 3: padded segment offsets + tile->expert map ----------------

__global__ void offsets_kernel(const int* __restrict__ cnt, int* __restrict__ cursor,
                               int* __restrict__ tileExpert) {
  int off[E + 1];
  off[0] = 0;
  for (int e = 0; e < E; ++e) {
    cursor[e] = off[e];
    const int padded = ((cnt[e] + TILE_M - 1) / TILE_M) * TILE_M;
    off[e + 1] = off[e] + padded;
  }
  const int nTiles = off[E] / TILE_M;
  int e = 0;
  for (int t = 0; t < MAX_TILES; ++t) {
    if (t < nTiles) {
      const int base = t * TILE_M;
      while (base >= off[e + 1]) ++e;
      tileExpert[t] = e;
    } else {
      tileExpert[t] = -1;
    }
  }
}

// ---------------- kernel 4: scatter token ids/weights into expert segments ----------------

__global__ void scatter_kernel(const int* __restrict__ tok_e, const float* __restrict__ tok_w,
                               int* __restrict__ cursor, int* __restrict__ seg_tok,
                               float* __restrict__ seg_w) {
  const int n = blockIdx.x * blockDim.x + threadIdx.x;
  if (n >= N) return;
#pragma unroll
  for (int k = 0; k < TOPK; ++k) {
    const int e = tok_e[2 * n + k];
    const int pos = atomicAdd(&cursor[e], 1);
    seg_tok[pos] = n;
    seg_w[pos] = tok_w[2 * n + k];
  }
}

// ---------------- kernel 5: grouped expert GEMM + GELU + weighted combine ----------------
// Block tile 128x256 (8 waves, each 64x64 = 4x4 WMMA C-tiles). Double-buffered LDS.
// Tile movement uses CDNA5 async global->LDS copies (ASYNCcnt) so no staging VGPRs
// exist: asyncs for buffer c^1 are issued before the WMMAs on buffer c, drained with
// s_wait_asynccnt 0 just before the single per-iteration barrier.
// Padding rows (token == -1) read garbage into A; row m of A only affects row m of C
// and the epilogue skips token<0 stores, so no zero-fill (and no divergence) needed.

__global__ __launch_bounds__(256, 2)
void moe_gemm_kernel(const u16* __restrict__ xb, const u16* __restrict__ wbT,
                     const float* __restrict__ eb, const int* __restrict__ seg_tok,
                     const float* __restrict__ seg_w, const int* __restrict__ tileExpert,
                     float* __restrict__ out) {
  __shared__ u16 As[2][TILE_M][LDS_STRIDE];    // gathered activation rows, [row][k]
  __shared__ u16 Bs[2][TILE_NB][LDS_STRIDE];   // weight columns, [col][k] (pre-transposed)

  const int tile = blockIdx.x;
  const int e = tileExpert[tile];
  if (e < 0) return;
  const int mBase = tile * TILE_M;
  const int hBase = blockIdx.y * TILE_NB;

  const int tid  = threadIdx.x;
  const int lane = tid & 31;
  const int wave = tid >> 5;
  const int wm = wave >> 2;                    // 0..1 : 64-row half
  const int wn = wave & 3;                     // 0..3 : 64-col quarter

  v8f acc[4][4] = {};

  // A loader: 2 threads per row, 16 bf16 (32B = 2 x b128) each
  const int arow  = tid >> 1;
  const int ahalf = tid & 1;
  const int atok  = seg_tok[mBase + arow];     // -1 for padding rows (garbage OK, see above)
  const u16* aSrc = xb + (size_t)(atok < 0 ? 0 : atok) * D + ahalf * 16;
  // B loader: 1 thread per column, 32 bf16 (64B = 4 x b128) each
  const int bcol = tid;
  const u16* bSrc = wbT + ((size_t)e * H + hBase + bcol) * D;

  auto issue_async = [&](int buf, int kb) {    // memory -> LDS, no VGPR data path
    async_copy_b128(&As[buf][arow][ahalf * 16],     aSrc + kb);
    async_copy_b128(&As[buf][arow][ahalf * 16 + 8], aSrc + kb + 8);
#pragma unroll
    for (int i = 0; i < 4; ++i)
      async_copy_b128(&Bs[buf][bcol][i * 8], bSrc + kb + i * 8);
    if (kb + TILE_K < D) {                     // stream-ahead prefetch (high locality)
      __builtin_prefetch(aSrc + kb + TILE_K, 0, 3);
      __builtin_prefetch(bSrc + kb + TILE_K, 0, 3);
    }
  };

  issue_async(0, 0);
  wait_asynccnt0();
  __syncthreads();

  const int kOff = (lane >> 4) * 8;            // ISA 16-bit operand lane split
  const int fr   = lane & 15;

  int buf = 0;
  for (int kb = 0; kb < D; kb += TILE_K, buf ^= 1) {
    const bool hasNext = (kb + TILE_K) < D;
    if (hasNext) issue_async(buf ^ 1, kb + TILE_K);  // overlap copies with WMMAs

    v16bf af[4], bfv[4];
#pragma unroll
    for (int tm = 0; tm < 4; ++tm)
      af[tm] = load_frag(&As[buf][wm * 64 + tm * 16 + fr][kOff]);
#pragma unroll
    for (int tn = 0; tn < 4; ++tn)
      bfv[tn] = load_frag(&Bs[buf][wn * 64 + tn * 16 + fr][kOff]);

#pragma unroll
    for (int tm = 0; tm < 4; ++tm)
#pragma unroll
      for (int tn = 0; tn < 4; ++tn)
        acc[tm][tn] = __builtin_amdgcn_wmma_f32_16x16x32_bf16(
            false, af[tm], false, bfv[tn], (short)0, acc[tm][tn], false, false);

    if (hasNext) wait_asynccnt0();             // async writes to buf^1 visible...
    __syncthreads();                           // ...to all waves after the barrier
  }

  // Epilogue: bias + exact GELU, scale by routing weight, atomic combine.
  // C layout (ISA 7.12.2): VGPR r -> M = r + 8*(lane>=16); N = lane&15.
  const int mSub = (lane >> 4) * 8;
#pragma unroll
  for (int tm = 0; tm < 4; ++tm) {
#pragma unroll
    for (int r = 0; r < 8; ++r) {
      const int grow = mBase + wm * 64 + tm * 16 + mSub + r;
      const int token = seg_tok[grow];
      if (token < 0) continue;
      const float w = seg_w[grow];
      float* orow = out + (size_t)token * H;
#pragma unroll
      for (int tn = 0; tn < 4; ++tn) {
        const int col = hBase + wn * 64 + tn * 16 + fr;
        const float v = acc[tm][tn][r] + eb[e * H + col];
        atomicAdd(orow + col, w * gelu_exact(v));
      }
    }
  }
}

// ---------------- launch ----------------

extern "C" void kernel_launch(void* const* d_in, const int* in_sizes, int n_in,
                              void* d_out, int out_size, void* d_ws, size_t ws_size,
                              hipStream_t stream) {
  (void)in_sizes; (void)n_in; (void)ws_size;
  const float* x  = (const float*)d_in[0];
  const float* rw = (const float*)d_in[1];
  const float* rb = (const float*)d_in[2];
  const float* ew = (const float*)d_in[3];
  const float* eb = (const float*)d_in[4];
  float* out = (float*)d_out;

  char* base = (char*)d_ws;
  size_t off = 0;
  auto take = [&](size_t bytes) -> void* {
    void* p = base + off;
    off = (off + bytes + 255) & ~(size_t)255;
    return p;
  };
  u16*   xb      = (u16*)  take((size_t)N * D * sizeof(u16));       // 50.3 MB
  u16*   wbT     = (u16*)  take((size_t)E * H * D * sizeof(u16));   // 18.9 MB, [E][H][D]
  int*   tok_e   = (int*)  take((size_t)N * TOPK * sizeof(int));
  float* tok_w   = (float*)take((size_t)N * TOPK * sizeof(float));
  int*   cnt     = (int*)  take(E * sizeof(int));
  int*   cursor  = (int*)  take(E * sizeof(int));
  int*   tileExp = (int*)  take(MAX_TILES * sizeof(int));
  int*   seg_tok = (int*)  take((size_t)MAX_ROWS * sizeof(int));
  float* seg_w   = (float*)take((size_t)MAX_ROWS * sizeof(float));

  // Per-call (graph-replay-safe) re-initialization.
  hipMemsetAsync(cnt, 0, E * sizeof(int), stream);
  hipMemsetAsync(seg_tok, 0xFF, (size_t)MAX_ROWS * sizeof(int), stream);  // -1 = padding row
  hipMemsetAsync(out, 0, (size_t)out_size * sizeof(float), stream);

  convert_bf16_kernel<<<2048, 256, 0, stream>>>(x, xb, N * D);
  transpose_w_kernel<<<dim3(H / 32, D / 32, E), dim3(32, 8), 0, stream>>>(ew, wbT);
  router_kernel<<<N / 8, 256, 0, stream>>>(x, rw, rb, tok_e, tok_w, cnt);
  offsets_kernel<<<1, 1, 0, stream>>>(cnt, cursor, tileExp);
  scatter_kernel<<<N / 256, 256, 0, stream>>>(tok_e, tok_w, cursor, seg_tok, seg_w);
  moe_gemm_kernel<<<dim3(MAX_TILES, H / TILE_NB), 256, 0, stream>>>(
      xb, wbT, eb, seg_tok, seg_w, tileExp, out);
}
